// EmotionConditionedGNN_20212116095632
// MI455X (gfx1250) — matrix operand order repaired and checked by reading, not compiled, not run on hardware
//
#include <hip/hip_runtime.h>

// ---------------------------------------------------------------------------
// EmotionConditionedGNN on MI455X (gfx1250, wave32).
// - adjacency MLP depends only on emo_idx in {0..3} -> 4-entry table per WG.
// - enc (36x32), out1 (32x32), out2 (32x20) GEMMs run batched through
//   v_wmma_f32_16x16x32_f16 with (batch*node) as the WMMA M dimension.
// - weights staged f16/column-major in LDS so B-fragments are single 32B
//   LDS vector reads; feat in/delta out are float4 coalesced.
// ---------------------------------------------------------------------------

typedef __attribute__((ext_vector_type(16))) _Float16 v16h;
typedef __attribute__((ext_vector_type(8)))  float    v8f;

#define Nn 5
#define Dd 20
#define Ee 16
#define Hh 32
#define Cc 4
#define WGB 16             // batch elements per workgroup
#define ROWS (WGB * Nn)    // 80 rows of (batch*node) space
#define NTILES (ROWS / 16) // 5 WMMA M-tiles
#define XS 66              // x tile stride (halves): padded K=64 + slack
#define PS 36              // h_pre / delta-stage stride (floats, 16B-aligned rows)
#define HS 34              // h / h_agg / t stride (halves)
#define KPE 64             // staged enc_w K-pad
#define KPO 32             // staged out_w K-pad
#define NTHREADS 128
#define NWAVES (NTHREADS / 32)

// --- A-fragment (16x32 f16, M x K) from LDS row-major tile -----------------
// lane L holds M=L&15; lanes<16 K base 0, lanes>=16 base 16.
// half h -> k = kbase + h (contiguous, matches B-fragment K order).
__device__ __forceinline__ v16h load_afrag(const _Float16* base, int row0,
                                           int stride, int kc, int lane) {
  const int m  = lane & 15;
  const int kb = kc + ((lane & 16) ? 16 : 0);
  const _Float16* p = base + (row0 + m) * stride;
  v16h a;
#pragma unroll
  for (int h = 0; h < 16; ++h) {
    const int k = kb + h;
    a[h] = p[k];
  }
  return a;
}

// --- B-fragment (32x16 f16, K x N) from LDS column-major staged weight -----
// lane L holds column N = n0+(L&15); K halves kb..kb+15 are contiguous in
// wT[n*KP + k] -> one 32-byte LDS vector read.
__device__ __forceinline__ v16h load_bfrag_lds(const _Float16* wT, int KP,
                                               int kc, int n0, int lane) {
  const int n  = n0 + (lane & 15);
  const int kb = kc + ((lane & 16) ? 16 : 0);
  return *(const v16h*)&wT[n * KP + kb];
}

#define WMMA_F16(A, Bm, Cv) \
  __builtin_amdgcn_wmma_f32_16x16x32_f16(false, (A), false, (Bm), (short)0, (Cv), false, false)

__global__ __launch_bounds__(NTHREADS) void emo_gnn_kernel(
    const float* __restrict__ feat,      // (B,5,20)
    const int*   __restrict__ emo_idx,   // (B,)
    const float* __restrict__ emo_emb,   // (4,16)
    const float* __restrict__ A_w1, const float* __restrict__ A_b1,
    const float* __restrict__ A_w2, const float* __restrict__ A_b2,
    const float* __restrict__ enc_w, const float* __restrict__ enc_b,
    const float* __restrict__ ln_g,  const float* __restrict__ ln_b,
    const float* __restrict__ out_w1, const float* __restrict__ out_b1,
    const float* __restrict__ out_w2, const float* __restrict__ out_b2,
    const float* __restrict__ cls_w,  const float* __restrict__ cls_b,
    float* __restrict__ out, int Btot)
{
  __shared__ float                 sA[Cc * 25];     // softmaxed adjacency/emotion
  __shared__ int                   sEi[WGB];
  __shared__ __align__(4) _Float16 sEh[Cc * Ee];    // f16 emotion embeddings
  __shared__ __align__(32) _Float16 wEncT[Hh * KPE]; // enc_w^T, f16, K-padded
  __shared__ __align__(32) _Float16 wO1T[Hh * KPO];  // out_w1^T
  __shared__ __align__(32) _Float16 wO2T[Hh * KPO];  // out_w2^T (N-padded 20->32)
  __shared__ __align__(16) _Float16 sX[ROWS * XS];   // [feat ; e ; 0-pad] f16
  __shared__ __align__(16) float    sHpre[ROWS * PS]; // pre-LN enc out, then delta stage
  __shared__ _Float16 sH[ROWS * HS];                 // relu(LN(.))
  __shared__ _Float16 sAgg[ROWS * HS];               // A @ h
  __shared__ _Float16 sT[ROWS * HS];                 // relu(h_agg @ W1 + b1)

  const int tid  = threadIdx.x;
  const int lane = tid & 31;
  const int wv   = tid >> 5;
  const int b0   = blockIdx.x * WGB;

  float* outDelta = out;
  float* outAux   = out + (size_t)Btot * (Nn * Dd);
  float* outA     = out + (size_t)Btot * (Nn * Dd + Cc);

  // ---- phase 0: indices, f16 embeddings, adjacency table, weight staging ----
  if (tid < WGB) sEi[tid] = emo_idx[b0 + tid];
  if (tid < Cc * Ee) sEh[tid] = (_Float16)emo_emb[tid];
  if (tid < Cc) {
    const int c = tid;
    float e[Ee];
#pragma unroll
    for (int k = 0; k < Ee; ++k) e[k] = emo_emb[c * Ee + k];
    float arow[25];
#pragma unroll
    for (int t = 0; t < 25; ++t) arow[t] = A_b2[t];
    for (int j = 0; j < 32; ++j) {
      float s = A_b1[j];
#pragma unroll
      for (int k = 0; k < Ee; ++k) s += e[k] * A_w1[k * 32 + j];
      s = fmaxf(s, 0.0f);
#pragma unroll
      for (int t = 0; t < 25; ++t) arow[t] += s * A_w2[j * 25 + t];
    }
#pragma unroll
    for (int i = 0; i < Nn; ++i) {
      float mx = arow[i * 5];
#pragma unroll
      for (int j = 1; j < Nn; ++j) mx = fmaxf(mx, arow[i * 5 + j]);
      float sum = 0.0f;
#pragma unroll
      for (int j = 0; j < Nn; ++j) {
        const float ex = __expf(arow[i * 5 + j] - mx);
        arow[i * 5 + j] = ex;
        sum += ex;
      }
      const float inv = 1.0f / sum;
#pragma unroll
      for (int j = 0; j < Nn; ++j) sA[c * 25 + i * 5 + j] = arow[i * 5 + j] * inv;
    }
  }
  // stage f16 transposed weights: wT[n*KP + k] = W[k*Ncols + n] (0-padded)
  for (int idx = tid; idx < Hh * KPE; idx += NTHREADS) {
    const int n = idx / KPE, k = idx - n * KPE;
    wEncT[idx] = (_Float16)((k < Dd + Ee) ? enc_w[k * Hh + n] : 0.0f);
  }
  for (int idx = tid; idx < Hh * KPO; idx += NTHREADS) {
    const int n = idx / KPO, k = idx - n * KPO;
    wO1T[idx] = (_Float16)out_w1[k * Hh + n];
    wO2T[idx] = (_Float16)((n < Dd) ? out_w2[k * Dd + n] : 0.0f);
  }
  // prefetch next workgroup's feat stream into cache (global_prefetch_b8)
  if (b0 + WGB < Btot && tid < 50)
    __builtin_prefetch(feat + (size_t)(b0 + WGB) * (Nn * Dd) + tid * 32, 0, 1);
  __syncthreads();

  // ---- per-wave constant weight fragments (32B LDS vector reads) ----
  v16h wEnc[2][2], wO1[2], wO2[2];
#pragma unroll
  for (int kc = 0; kc < 2; ++kc)
#pragma unroll
    for (int nt = 0; nt < 2; ++nt)
      wEnc[kc][nt] = load_bfrag_lds(wEncT, KPE, kc * 32, nt * 16, lane);
  wO1[0] = load_bfrag_lds(wO1T, KPO, 0, 0,  lane);
  wO1[1] = load_bfrag_lds(wO1T, KPO, 0, 16, lane);
  wO2[0] = load_bfrag_lds(wO2T, KPO, 0, 0,  lane);
  wO2[1] = load_bfrag_lds(wO2T, KPO, 0, 16, lane);

  const int nlo = lane & 15;
  const int nhi = nlo + 16;
  const float encB0 = enc_b[nlo],  encB1 = enc_b[nhi];
  const float o1B0  = out_b1[nlo], o1B1  = out_b1[nhi];
  const float o2B0  = out_b2[nlo];                      // nlo < 20 always
  const float o2B1  = (nhi < Dd) ? out_b2[nhi] : 0.0f;

  // ---- phase 1: build x = [feat ; e ; 0] tile (f16) + emit A output ----
  for (int idx = tid; idx < ROWS * 5; idx += NTHREADS) {   // feat: float4 chunks
    const int r = idx / 5, q = idx - r * 5;
    const int bl = r / Nn, i = r - bl * Nn;
    const float4 f =
        *(const float4*)(feat + ((size_t)(b0 + bl) * Nn + i) * Dd + 4 * q);
    union { _Float16 h[4]; unsigned int u[2]; } cv;
    cv.h[0] = (_Float16)f.x; cv.h[1] = (_Float16)f.y;
    cv.h[2] = (_Float16)f.z; cv.h[3] = (_Float16)f.w;
    unsigned int* dst = (unsigned int*)&sX[r * XS + 4 * q];
    dst[0] = cv.u[0]; dst[1] = cv.u[1];
  }
  for (int idx = tid; idx < ROWS * 8; idx += NTHREADS) {   // e: dword copies
    const int r = idx >> 3, j = idx & 7;
    const int bl = r / Nn;
    const unsigned int v = *(const unsigned int*)&sEh[sEi[bl] * Ee + 2 * j];
    *(unsigned int*)&sX[r * XS + Dd + 2 * j] = v;
  }
  for (int idx = tid; idx < ROWS * 14; idx += NTHREADS) {  // zero pad cols 36..63
    const int r = idx / 14, j = idx - r * 14;
    *(unsigned int*)&sX[r * XS + (Dd + Ee) + 2 * j] = 0u;
  }
  for (int idx = tid; idx < WGB * 25 / 4; idx += NTHREADS) { // A out: float4
    float4 v;
    float* vp = (float*)&v;
#pragma unroll
    for (int j = 0; j < 4; ++j) {
      const int z4 = idx * 4 + j;
      const int bl = z4 / 25, z = z4 - bl * 25;
      vp[j] = sA[sEi[bl] * 25 + z];
    }
    *(float4*)(outA + (size_t)b0 * 25 + idx * 4) = v;
  }
  __syncthreads();

  // ---- phase 2: encoder GEMM via WMMA (K=64 padded: 2 chunks x 2 N-tiles) ----
  for (int t = wv; t < NTILES; t += NWAVES) {
    const int row0 = t * 16;
    const v16h a0 = load_afrag(sX, row0, XS, 0,  lane);
    const v16h a1 = load_afrag(sX, row0, XS, 32, lane);
    v8f c0 = {}, c1 = {};
    c0 = WMMA_F16(a0, wEnc[0][0], c0);
    c0 = WMMA_F16(a1, wEnc[1][0], c0);
    c1 = WMMA_F16(a0, wEnc[0][1], c1);
    c1 = WMMA_F16(a1, wEnc[1][1], c1);
    const int mb = (lane & 16) ? 8 : 0;
#pragma unroll
    for (int r = 0; r < 8; ++r) {
      const int m = row0 + mb + r;
      sHpre[m * PS + nlo] = c0[r] + encB0;
      sHpre[m * PS + nhi] = c1[r] + encB1;
    }
  }
  __syncthreads();

  // ---- phase 3: LayerNorm + ReLU (one thread per row) ----
  if (tid < ROWS) {
    const float* p = &sHpre[tid * PS];
    float mu = 0.0f, m2 = 0.0f;
#pragma unroll
    for (int c = 0; c < Hh; ++c) { const float v = p[c]; mu += v; m2 += v * v; }
    mu *= (1.0f / Hh);
    const float var = m2 * (1.0f / Hh) - mu * mu;
    const float rs = rsqrtf(var + 1e-5f);
#pragma unroll
    for (int c = 0; c < Hh; ++c) {
      const float v = (p[c] - mu) * rs * ln_g[c] + ln_b[c];
      sH[tid * HS + c] = (_Float16)fmaxf(v, 0.0f);
    }
  }
  __syncthreads();

  // ---- phase 4: graph aggregation h_agg = A @ h (5-term dots in VALU) ----
  for (int idx = tid; idx < ROWS * Hh; idx += NTHREADS) {
    const int r = idx >> 5, n = idx & 31;
    const int bl = r / Nn, i = r - bl * Nn;
    const float* Ar = &sA[sEi[bl] * 25 + i * 5];
    float acc = 0.0f;
#pragma unroll
    for (int j = 0; j < Nn; ++j)
      acc += Ar[j] * (float)sH[(bl * Nn + j) * HS + n];
    sAgg[r * HS + n] = (_Float16)acc;
  }
  __syncthreads();

  // ---- phase 5: out MLP via WMMA. Uniform 2-iteration loop so the
  //      sT write->read hand-off can use a non-divergent __syncthreads().
  //      Stage-3 result goes to sHpre (free after LN) for coalesced writeout.
#pragma unroll 1
  for (int tt = 0; tt < 2; ++tt) {
    const int t = wv + tt * NWAVES;
    const int row0 = t * 16;
    if (t < NTILES) {
      const v16h a = load_afrag(sAgg, row0, HS, 0, lane);
      v8f c0 = {}, c1 = {};
      c0 = WMMA_F16(a, wO1[0], c0);
      c1 = WMMA_F16(a, wO1[1], c1);
      const int mb = (lane & 16) ? 8 : 0;
#pragma unroll
      for (int r = 0; r < 8; ++r) {
        const int m = row0 + mb + r;
        sT[m * HS + nlo] = (_Float16)fmaxf(c0[r] + o1B0, 0.0f);
        sT[m * HS + nhi] = (_Float16)fmaxf(c1[r] + o1B1, 0.0f);
      }
    }
    __syncthreads();
    if (t < NTILES) {
      const v16h a2 = load_afrag(sT, row0, HS, 0, lane);
      v8f d0 = {}, d1 = {};
      d0 = WMMA_F16(a2, wO2[0], d0);
      d1 = WMMA_F16(a2, wO2[1], d1);
      const int mb = (lane & 16) ? 8 : 0;
#pragma unroll
      for (int r = 0; r < 8; ++r) {
        const int m = row0 + mb + r;
        sHpre[m * PS + nlo] = d0[r] + o2B0;
        if (nhi < Dd) sHpre[m * PS + nhi] = d1[r] + o2B1;
      }
    }
  }
  __syncthreads();

  // ---- phase 6: coalesced float4 delta writeout + classifier logits ----
  for (int idx = tid; idx < ROWS * 5; idx += NTHREADS) {
    const int r = idx / 5, q = idx - r * 5;
    const float4 v = *(const float4*)&sHpre[r * PS + 4 * q];
    *(float4*)(outDelta + (size_t)b0 * (Nn * Dd) + r * Dd + 4 * q) = v;
  }
  if (tid < WGB * Cc) {
    const int bl = tid >> 2, c = tid & 3;
    float acc = cls_b[c];
#pragma unroll
    for (int i = 0; i < Nn; ++i)
      for (int n = 0; n < Hh; ++n)
        acc += (float)sAgg[(bl * Nn + i) * HS + n] * cls_w[(i * Hh + n) * Cc + c];
    outAux[(size_t)(b0 + bl) * Cc + c] = acc;
  }
}

extern "C" void kernel_launch(void* const* d_in, const int* in_sizes, int n_in,
                              void* d_out, int out_size, void* d_ws, size_t ws_size,
                              hipStream_t stream) {
  (void)n_in; (void)out_size; (void)d_ws; (void)ws_size;
  const float* feat    = (const float*)d_in[0];
  const int*   emo_idx = (const int*)  d_in[1];
  const float* emo_emb = (const float*)d_in[2];
  const float* A_w1    = (const float*)d_in[3];
  const float* A_b1    = (const float*)d_in[4];
  const float* A_w2    = (const float*)d_in[5];
  const float* A_b2    = (const float*)d_in[6];
  const float* enc_w   = (const float*)d_in[7];
  const float* enc_b   = (const float*)d_in[8];
  const float* ln_g    = (const float*)d_in[9];
  const float* ln_b    = (const float*)d_in[10];
  const float* out_w1  = (const float*)d_in[11];
  const float* out_b1  = (const float*)d_in[12];
  const float* out_w2  = (const float*)d_in[13];
  const float* out_b2  = (const float*)d_in[14];
  const float* cls_w   = (const float*)d_in[15];
  const float* cls_b   = (const float*)d_in[16];

  const int Btot = in_sizes[1];         // emo_idx: one entry per batch element
  const int grid = Btot / WGB;

  emo_gnn_kernel<<<grid, NTHREADS, 0, stream>>>(
      feat, emo_idx, emo_emb, A_w1, A_b1, A_w2, A_b2,
      enc_w, enc_b, ln_g, ln_b, out_w1, out_b1, out_w2, out_b2,
      cls_w, cls_b, (float*)d_out, Btot);
}
